// MultiHeadAttention_46660524703774
// MI455X (gfx1250) — matrix operand order repaired and checked by reading, not compiled, not run on hardware
//
#include <hip/hip_runtime.h>
#include <hip/hip_bf16.h>

typedef __attribute__((ext_vector_type(16))) _Float16 v16h;
typedef __attribute__((ext_vector_type(8)))  _Float16 v8h;
typedef __attribute__((ext_vector_type(8)))  float    v8f;

#define B_  2
#define L_  2048
#define D_  1024
#define H_  16
#define DH_ 64

static __device__ __forceinline__ v16h cat16(v8h a, v8h b) {
  return __builtin_shufflevector(a, b, 0,1,2,3,4,5,6,7,8,9,10,11,12,13,14,15);
}

static __device__ __forceinline__ v8f wmma_f16f32(v16h a, v16h b, v8f c) {
  // D = A(16x32 f16) * B(32x16 f16) + C(16x16 f32)
  return __builtin_amdgcn_wmma_f32_16x16x32_f16(false, a, false, b, (short)0, c, false, false);
}

// Async memory->LDS copy, 16 bytes per lane, tracked by ASYNCcnt.
// GVS mode: mem = SADDR(u64) + VADDR(i32) + IOFFSET. LDS addr in VGPR %0.
static __device__ __forceinline__ void async_load16(unsigned lds_off, unsigned gbyte_off,
                                                    const void* base) {
  asm volatile("global_load_async_to_lds_b128 %0, %1, %2 offset:0"
               :
               : "v"(lds_off), "v"(gbyte_off), "s"(base)
               : "memory");
}
static __device__ __forceinline__ void wait_async_le4() {
  asm volatile("s_wait_asynccnt 0x4" ::: "memory");
}
static __device__ __forceinline__ void wait_async_le2() {
  asm volatile("s_wait_asynccnt 0x2" ::: "memory");
}
static __device__ __forceinline__ void wait_async_0() {
  asm volatile("s_wait_asynccnt 0x0" ::: "memory");
}

static __device__ __forceinline__ float block_reduce_sum_256(float v, float* sbuf) {
  const int t = threadIdx.x;
  sbuf[t] = v;
  __syncthreads();
  for (int s = 128; s >= 1; s >>= 1) {
    if (t < s) sbuf[t] += sbuf[t + s];
    __syncthreads();
  }
  float r = sbuf[0];
  __syncthreads();
  return r;
}

// ---------------------------------------------------------------------------
// f32 -> f16 weight conversion
// ---------------------------------------------------------------------------
__global__ void cvt_f32_to_f16(const float* __restrict__ src, _Float16* __restrict__ dst, int n) {
  int i = blockIdx.x * 256 + threadIdx.x;
  if (i < n) dst[i] = (_Float16)src[i];
}

// ---------------------------------------------------------------------------
// LayerNorm(x) -> f16 activations (block per token, 256 threads, D=1024)
// ---------------------------------------------------------------------------
__global__ __launch_bounds__(256)
void ln_x_f16(const float* __restrict__ x, const float* __restrict__ w,
              const float* __restrict__ bia, _Float16* __restrict__ out) {
  __shared__ float sbuf[256];
  const int tok = blockIdx.x;
  const float* xr = x + (size_t)tok * D_;
  float vals[4];
  float s = 0.f, sq = 0.f;
#pragma unroll
  for (int k = 0; k < 4; k++) {
    int i = threadIdx.x + k * 256;
    float v = xr[i];
    vals[k] = v; s += v; sq += v * v;
  }
  s  = block_reduce_sum_256(s, sbuf);
  sq = block_reduce_sum_256(sq, sbuf);
  float mu = s * (1.0f / D_);
  float var = sq * (1.0f / D_) - mu * mu;
  float rs = rsqrtf(var + 1e-5f);
#pragma unroll
  for (int k = 0; k < 4; k++) {
    int i = threadIdx.x + k * 256;
    out[(size_t)tok * D_ + i] = (_Float16)((vals[k] - mu) * rs * w[i] + bia[i]);
  }
}

// ---------------------------------------------------------------------------
// C[M,N](f32) = A[M,K](f16, row-major) * W[N,K](f16, row-major)^T
// 128x128x32 tiles, async double-buffered LDS staging, 8 waves per block,
// each wave owns a 32x64 patch (2x4 accumulator tiles).
// ---------------------------------------------------------------------------
__global__ __launch_bounds__(256)
void wmma_gemm_nt(const _Float16* __restrict__ A, const _Float16* __restrict__ W,
                  float* __restrict__ C, int M, int N, int K) {
  __shared__ __align__(16) _Float16 As[2][128 * 32];
  __shared__ __align__(16) _Float16 Bs[2][128 * 32];
  const int t    = threadIdx.x;
  const int lane = t & 31;
  const int wid  = t >> 5;
  const int wm   = wid >> 1;     // 0..3  -> 32-row strip
  const int wn   = wid & 1;      // 0..1  -> 64-col strip
  const int lr   = lane & 15;
  const int hl   = (lane < 16) ? 0 : 1;
  const int rowBase = blockIdx.y * 128;
  const int colBase = blockIdx.x * 128;

  // Per-thread slice of the cooperative tile fill: 2 x 16B per matrix.
  const int r0  = t >> 2;                    // 0..63
  const int r1  = r0 + 64;                   // 64..127
  const int ch0 = (t & 3) * 8;               // half-index within 32-wide row

  auto issue_tile = [&](int buf, int kt) {
    async_load16((unsigned)(uintptr_t)&As[buf][r0 * 32 + ch0],
                 (unsigned)(((size_t)(rowBase + r0) * K + kt + ch0) * 2), A);
    async_load16((unsigned)(uintptr_t)&Bs[buf][r0 * 32 + ch0],
                 (unsigned)(((size_t)(colBase + r0) * K + kt + ch0) * 2), W);
    async_load16((unsigned)(uintptr_t)&As[buf][r1 * 32 + ch0],
                 (unsigned)(((size_t)(rowBase + r1) * K + kt + ch0) * 2), A);
    async_load16((unsigned)(uintptr_t)&Bs[buf][r1 * 32 + ch0],
                 (unsigned)(((size_t)(colBase + r1) * K + kt + ch0) * 2), W);
  };

  const v8f vzero = {0.f, 0.f, 0.f, 0.f, 0.f, 0.f, 0.f, 0.f};
  v8f acc[2][4];
#pragma unroll
  for (int i = 0; i < 2; i++)
#pragma unroll
    for (int j = 0; j < 4; j++) acc[i][j] = vzero;

  issue_tile(0, 0);
  const int nk = K >> 5;
  for (int it = 0; it < nk; it++) {
    const int cur = it & 1;
    __syncthreads();                         // prev compute done before overwriting nxt buffer
    if (it + 1 < nk) {
      issue_tile(cur ^ 1, (it + 1) << 5);    // prefetch next tile
      wait_async_le4();                      // this iteration's 4 loads complete (in-order)
    } else {
      wait_async_0();
    }
    __syncthreads();                         // all waves' tile data visible

    v16h af[2];
#pragma unroll
    for (int r2 = 0; r2 < 2; r2++) {
      int row = wm * 32 + r2 * 16 + lr;
      int lo  = hl * 8;
      v8h p0 = *(const v8h*)&As[cur][row * 32 + lo];
      v8h p1 = *(const v8h*)&As[cur][row * 32 + lo + 16];
      af[r2] = cat16(p0, p1);
    }
    v16h bf[4];
#pragma unroll
    for (int c2 = 0; c2 < 4; c2++) {
      int n  = wn * 64 + c2 * 16 + lr;
      int k0 = hl * 8;
      v8h p0 = *(const v8h*)&Bs[cur][n * 32 + k0];
      v8h p1 = *(const v8h*)&Bs[cur][n * 32 + k0 + 16];
      bf[c2] = cat16(p0, p1);
    }
#pragma unroll
    for (int r2 = 0; r2 < 2; r2++)
#pragma unroll
      for (int c2 = 0; c2 < 4; c2++)
        acc[r2][c2] = wmma_f16f32(af[r2], bf[c2], acc[r2][c2]);
  }

#pragma unroll
  for (int r2 = 0; r2 < 2; r2++)
#pragma unroll
    for (int c2 = 0; c2 < 4; c2++)
#pragma unroll
      for (int j = 0; j < 8; j++) {
        int gr = rowBase + wm * 32 + r2 * 16 + j + hl * 8;
        int gc = colBase + wn * 64 + c2 * 16 + lr;
        C[(size_t)gr * N + gc] = acc[r2][c2][j];
      }
}

// ---------------------------------------------------------------------------
// Per-token: weight-only LN on q,k + RoPE + pack to f16 heads.
// q,k -> [B*H, L, DH];  v -> transposed [B*H, DH, L] (for contiguous PV frags)
// ---------------------------------------------------------------------------
__global__ __launch_bounds__(256)
void qk_ln_rope_pack(const float* __restrict__ qkv, const float* __restrict__ qw,
                     const float* __restrict__ kw, _Float16* __restrict__ qp,
                     _Float16* __restrict__ kp, _Float16* __restrict__ vtp) {
  __shared__ float sbuf[256];
  const int tok = blockIdx.x;
  const int b   = tok >> 11;        // /L_
  const int pos = tok & (L_ - 1);
  const float* qr = qkv + (size_t)tok * (3 * D_);
  const float* kr = qr + D_;
  const float* vr = qr + 2 * D_;

  float qs = 0.f, qq = 0.f, ks = 0.f, kq = 0.f;
#pragma unroll
  for (int kk = 0; kk < 4; kk++) {
    int i = threadIdx.x + kk * 256;
    float q = qr[i], k = kr[i];
    qs += q; qq += q * q; ks += k; kq += k * k;
  }
  qs = block_reduce_sum_256(qs, sbuf);
  qq = block_reduce_sum_256(qq, sbuf);
  ks = block_reduce_sum_256(ks, sbuf);
  kq = block_reduce_sum_256(kq, sbuf);
  float qmu = qs * (1.0f / D_);
  float qrs = rsqrtf(qq * (1.0f / D_) - qmu * qmu + 1e-5f);
  float kmu = ks * (1.0f / D_);
  float krs = rsqrtf(kq * (1.0f / D_) - kmu * kmu + 1e-5f);

  // RoPE over pairs (d, d+32) within each 64-dim head
#pragma unroll
  for (int kk = 0; kk < 2; kk++) {
    int p  = threadIdx.x + kk * 256;  // 0..511 pairs
    int hh = p >> 5;
    int d  = p & 31;
    int i1 = hh * 64 + d;
    int i2 = i1 + 32;
    float q1 = (qr[i1] - qmu) * qrs * qw[i1];
    float q2 = (qr[i2] - qmu) * qrs * qw[i2];
    float k1 = (kr[i1] - kmu) * krs * kw[i1];
    float k2 = (kr[i2] - kmu) * krs * kw[i2];
    // inv_freq = 10000^(-d/32) = exp(-d * ln(10000)/32)
    float ang = (float)pos * __expf(-(float)d * (9.210340371976184f / 32.0f));
    float s, c;
    __sincosf(ang, &s, &c);
    size_t base = ((size_t)(b * H_ + hh) * L_ + pos) * DH_ + d;
    qp[base]      = (_Float16)(q1 * c - q2 * s);
    qp[base + 32] = (_Float16)(q2 * c + q1 * s);
    kp[base]      = (_Float16)(k1 * c - k2 * s);
    kp[base + 32] = (_Float16)(k2 * c + k1 * s);
  }
  // V transpose pack
#pragma unroll
  for (int kk = 0; kk < 4; kk++) {
    int i  = threadIdx.x + kk * 256;
    int hh = i >> 6;
    int d  = i & 63;
    vtp[((size_t)(b * H_ + hh) * DH_ + d) * L_ + pos] = (_Float16)vr[i];
  }
}

// ---------------------------------------------------------------------------
// Flash attention: grid (B*H, L/128), 8 waves x 16 query rows per block.
// Async double-buffered K (32x64) and V^T (64x32) tiles; online softmax.
// ---------------------------------------------------------------------------
__global__ __launch_bounds__(256)
void flash_attn(const _Float16* __restrict__ qp, const _Float16* __restrict__ kp,
                const _Float16* __restrict__ vtp, _Float16* __restrict__ ctx) {
  __shared__ __align__(16) _Float16 Kt[2][32 * 64];
  __shared__ __align__(16) _Float16 Vt[2][64 * 32];
  __shared__ __align__(16) _Float16 Ps[8][16 * 32];

  const int t    = threadIdx.x;
  const int lane = t & 31;
  const int wid  = t >> 5;
  const int bh   = blockIdx.x;          // 0..31
  const int b    = bh >> 4;
  const int h    = bh & (H_ - 1);
  const int lr   = lane & 15;
  const int hl   = (lane < 16) ? 0 : 1;
  const int q0   = blockIdx.y * 128 + wid * 16;

  // Per-thread slice of cooperative K/V tile fill (16B each).
  const int kr_  = t >> 3, kc_ = (t & 7) * 8;   // K: 32 rows x 8 chunks
  const int vr_  = t >> 2, vc_ = (t & 3) * 8;   // Vt: 64 rows x 4 chunks

  auto issue_tiles = [&](int buf, int kb) {
    async_load16((unsigned)(uintptr_t)&Kt[buf][kr_ * 64 + kc_],
                 (unsigned)((((size_t)bh * L_ + kb + kr_) * DH_ + kc_) * 2), kp);
    async_load16((unsigned)(uintptr_t)&Vt[buf][vr_ * 32 + vc_],
                 (unsigned)((((size_t)bh * DH_ + vr_) * L_ + kb + vc_) * 2), vtp);
  };

  // Q A-fragments (16x32 each; frag0 dims 0..31, frag1 dims 32..63)
  v16h qa[2];
  {
    const _Float16* qrow = qp + ((size_t)bh * L_ + q0 + lr) * DH_;
    int lo = hl * 8;
#pragma unroll
    for (int f = 0; f < 2; f++) {
      v8h p0 = *(const v8h*)&qrow[f * 32 + lo];
      v8h p1 = *(const v8h*)&qrow[f * 32 + lo + 16];
      qa[f] = cat16(p0, p1);
    }
  }

  const v8f vzero = {0.f, 0.f, 0.f, 0.f, 0.f, 0.f, 0.f, 0.f};
  float mrow[8], srow[8];
  v8f oacc[4];
#pragma unroll
  for (int j = 0; j < 8; j++) { mrow[j] = -1e30f; srow[j] = 0.f; }
#pragma unroll
  for (int dtc = 0; dtc < 4; dtc++) oacc[dtc] = vzero;

  issue_tiles(0, 0);
  const int nkb = L_ >> 5;
  for (int it = 0; it < nkb; it++) {
    const int cur = it & 1;
    __syncthreads();                        // prev compute done before overwriting nxt buffer
    if (it + 1 < nkb) {
      issue_tiles(cur ^ 1, (it + 1) << 5);  // prefetch next K/V tiles
      wait_async_le2();
    } else {
      wait_async_0();
    }
    __syncthreads();                        // tile data visible to all waves

    // S = (Q K^T) * 1/sqrt(DH), two 16x16 tiles covering 32 keys
    v8f st[2];
#pragma unroll
    for (int t2 = 0; t2 < 2; t2++) {
      int n  = t2 * 16 + lr;
      int k0 = hl * 8;
      v8h a0 = *(const v8h*)&Kt[cur][n * 64 + k0];
      v8h a1 = *(const v8h*)&Kt[cur][n * 64 + k0 + 16];
      v8h b0 = *(const v8h*)&Kt[cur][n * 64 + 32 + k0];
      v8h b1 = *(const v8h*)&Kt[cur][n * 64 + 32 + k0 + 16];
      v8f c = vzero;
      c = wmma_f16f32(qa[0], cat16(a0, a1), c);
      c = wmma_f16f32(qa[1], cat16(b0, b1), c);
      st[t2] = c * 0.125f;
    }

    // online softmax (row j lives across 16 lanes of this half-wave)
#pragma unroll
    for (int j = 0; j < 8; j++) {
      float mx = fmaxf(st[0][j], st[1][j]);
#pragma unroll
      for (int off = 1; off < 16; off <<= 1) mx = fmaxf(mx, __shfl_xor(mx, off, 32));
      float mn = fmaxf(mrow[j], mx);
      float sc = __expf(mrow[j] - mn);
      float p0 = __expf(st[0][j] - mn);
      float p1 = __expf(st[1][j] - mn);
      float sm = p0 + p1;
#pragma unroll
      for (int off = 1; off < 16; off <<= 1) sm += __shfl_xor(sm, off, 32);
      srow[j] = srow[j] * sc + sm;
      mrow[j] = mn;
#pragma unroll
      for (int dtc = 0; dtc < 4; dtc++) oacc[dtc][j] *= sc;
      // stage P (C-layout -> LDS row-major 16x32)
      int prow = j + hl * 8;
      Ps[wid][prow * 32 + lr]      = (_Float16)p0;
      Ps[wid][prow * 32 + 16 + lr] = (_Float16)p1;
    }

    // reload P as A-fragment (16 queries x 32 keys)
    v16h pa;
    {
      int lo = hl * 8;
      v8h p0 = *(const v8h*)&Ps[wid][lr * 32 + lo];
      v8h p1 = *(const v8h*)&Ps[wid][lr * 32 + lo + 16];
      pa = cat16(p0, p1);
    }
    // O += P * V  (4 output tiles over DH=64)
#pragma unroll
    for (int dtc = 0; dtc < 4; dtc++) {
      int n  = dtc * 16 + lr;
      int k0 = hl * 8;
      v8h p0 = *(const v8h*)&Vt[cur][n * 32 + k0];
      v8h p1 = *(const v8h*)&Vt[cur][n * 32 + k0 + 16];
      oacc[dtc] = wmma_f16f32(pa, cat16(p0, p1), oacc[dtc]);
    }
  }

  // normalize + store ctx as f16 token-major [B*L, D]
#pragma unroll
  for (int dtc = 0; dtc < 4; dtc++)
#pragma unroll
    for (int j = 0; j < 8; j++) {
      int gl = q0 + j + hl * 8;
      float val = oacc[dtc][j] / srow[j];
      ctx[((size_t)b * L_ + gl) * D_ + h * DH_ + dtc * 16 + lr] = (_Float16)val;
    }
}

// ---------------------------------------------------------------------------
extern "C" void kernel_launch(void* const* d_in, const int* in_sizes, int n_in,
                              void* d_out, int out_size, void* d_ws, size_t ws_size,
                              hipStream_t stream) {
  (void)in_sizes; (void)n_in; (void)out_size; (void)ws_size;
  const float* x      = (const float*)d_in[0];
  const float* ln_w   = (const float*)d_in[1];
  const float* ln_b   = (const float*)d_in[2];
  const float* w_qkv  = (const float*)d_in[3];
  const float* q_ln_w = (const float*)d_in[4];
  const float* k_ln_w = (const float*)d_in[5];
  const float* w_out  = (const float*)d_in[6];
  float* out = (float*)d_out;

  char* ws = (char*)d_ws;
  size_t off = 0;
  auto alloc = [&](size_t bytes) -> void* {
    void* p = ws + off;
    off += (bytes + 255) & ~(size_t)255;
    return p;
  };
  const size_t NT = (size_t)B_ * L_;                       // 4096 tokens
  _Float16* hA     = (_Float16*)alloc(NT * D_ * 2);        // LN(x) f16
  _Float16* wqkv16 = (_Float16*)alloc((size_t)3 * D_ * D_ * 2);
  _Float16* wout16 = (_Float16*)alloc((size_t)D_ * D_ * 2);
  float*    qkv    = (float*)   alloc(NT * 3 * D_ * 4);
  _Float16* qp     = (_Float16*)alloc((size_t)B_ * H_ * L_ * DH_ * 2);
  _Float16* kp     = (_Float16*)alloc((size_t)B_ * H_ * L_ * DH_ * 2);
  _Float16* vtp    = (_Float16*)alloc((size_t)B_ * H_ * DH_ * L_ * 2);
  _Float16* ctxb   = (_Float16*)alloc(NT * D_ * 2);

  {
    int n = 3 * D_ * D_;
    cvt_f32_to_f16<<<(n + 255) / 256, 256, 0, stream>>>(w_qkv, wqkv16, n);
  }
  {
    int n = D_ * D_;
    cvt_f32_to_f16<<<(n + 255) / 256, 256, 0, stream>>>(w_out, wout16, n);
  }
  ln_x_f16<<<(int)NT, 256, 0, stream>>>(x, ln_w, ln_b, hA);
  wmma_gemm_nt<<<dim3((3 * D_) / 128, (int)NT / 128), 256, 0, stream>>>(
      hA, wqkv16, qkv, (int)NT, 3 * D_, D_);
  qk_ln_rope_pack<<<(int)NT, 256, 0, stream>>>(qkv, q_ln_w, k_ln_w, qp, kp, vtp);
  flash_attn<<<dim3(B_ * H_, L_ / 128), 256, 0, stream>>>(qp, kp, vtp, ctxb);
  wmma_gemm_nt<<<dim3(D_ / 128, (int)NT / 128), 256, 0, stream>>>(
      ctxb, wout16, out, (int)NT, D_, D_);
}
